// SparseLinear_76295799046852
// MI455X (gfx1250) — matrix operand order repaired and checked by reading, not compiled, not run on hardware
//
#include <hip/hip_runtime.h>
#include <hip/hip_bf16.h>

// Problem sizes (fixed by the reference)
#define M_DIM 8192   // batch
#define N_DIM 4096   // out_features
#define K_DIM 4096   // in_features

typedef __attribute__((ext_vector_type(16))) __bf16 v16bf;
typedef __attribute__((ext_vector_type(8)))  __bf16 v8bf;
typedef __attribute__((ext_vector_type(8)))  float  v8f;
typedef __attribute__((ext_vector_type(4)))  float  v4f;
typedef __attribute__((ext_vector_type(4)))  int    v4i;

// ---- async global->LDS path (ASYNCcnt) with sync fallback -------------------
#if defined(__AMDGCN__) && __has_builtin(__builtin_amdgcn_global_load_async_to_lds_b128)
#define HAVE_ASYNC_LDS 1
#else
#define HAVE_ASYNC_LDS 0
#endif

#if HAVE_ASYNC_LDS
typedef __attribute__((address_space(1))) v4i gv4i_t;  // global int4
typedef __attribute__((address_space(3))) v4i lv4i_t;  // LDS int4
#endif

__device__ __forceinline__ void cp16_g2l(const __bf16* g, __bf16* l) {
#if HAVE_ASYNC_LDS
  __builtin_amdgcn_global_load_async_to_lds_b128(
      (gv4i_t*)g, (lv4i_t*)l, /*imm offset*/0, /*cpol*/0);
#else
  *(v8bf*)l = *(const v8bf*)g;
#endif
}

__device__ __forceinline__ void wait_async_done() {
#if HAVE_ASYNC_LDS
#if __has_builtin(__builtin_amdgcn_s_wait_asynccnt)
  __builtin_amdgcn_s_wait_asynccnt(0);
#else
  asm volatile("s_wait_asynccnt 0" ::: "memory");
#endif
#endif
}

// ---- pack kernels: f32 -> bf16 (mask folded into W) -------------------------
__global__ void pack_x_kernel(const float* __restrict__ x, __bf16* __restrict__ xb) {
  size_t i = ((size_t)blockIdx.x * blockDim.x + threadIdx.x) * 8;
  v4f a = *(const v4f*)(x + i);
  v4f b = *(const v4f*)(x + i + 4);
  v8bf o;
  o[0] = (__bf16)a[0]; o[1] = (__bf16)a[1]; o[2] = (__bf16)a[2]; o[3] = (__bf16)a[3];
  o[4] = (__bf16)b[0]; o[5] = (__bf16)b[1]; o[6] = (__bf16)b[2]; o[7] = (__bf16)b[3];
  *(v8bf*)(xb + i) = o;
}

__global__ void pack_w_kernel(const float* __restrict__ w,
                              const unsigned char* __restrict__ m,
                              __bf16* __restrict__ wb) {
  size_t i = ((size_t)blockIdx.x * blockDim.x + threadIdx.x) * 8;
  v4f a = *(const v4f*)(w + i);
  v4f b = *(const v4f*)(w + i + 4);
  const unsigned char* mp = m + i;
  v8bf o;
  o[0] = mp[0] ? (__bf16)a[0] : (__bf16)0.f;
  o[1] = mp[1] ? (__bf16)a[1] : (__bf16)0.f;
  o[2] = mp[2] ? (__bf16)a[2] : (__bf16)0.f;
  o[3] = mp[3] ? (__bf16)a[3] : (__bf16)0.f;
  o[4] = mp[4] ? (__bf16)b[0] : (__bf16)0.f;
  o[5] = mp[5] ? (__bf16)b[1] : (__bf16)0.f;
  o[6] = mp[6] ? (__bf16)b[2] : (__bf16)0.f;
  o[7] = mp[7] ? (__bf16)b[3] : (__bf16)0.f;
  *(v8bf*)(wb + i) = o;
}

// ---- main GEMM: Out[M,N] = Xb[M,K] * Wb[N,K]^T, bf16 in / f32 acc -----------
#define BM 128
#define BN 128
#define BK 32
#define NTHREADS 256   // 8 wave32 waves

__global__ __launch_bounds__(NTHREADS)
void gemm_bf16_wmma_kernel(const __bf16* __restrict__ Xb,
                           const __bf16* __restrict__ Wb,
                           float* __restrict__ Out) {
  __shared__ __bf16 As[2][BM][BK];   // 8 KB per stage
  __shared__ __bf16 Bs[2][BN][BK];   // 8 KB per stage

  const int tid   = threadIdx.x;
  const int lane  = tid & 31;
  const int wave  = tid >> 5;        // 0..7
  const int waveM = wave >> 1;       // 0..3 -> 32-row slab
  const int waveN = wave & 1;        // 0..1 -> 64-col slab
  const int l16   = lane & 15;
  const int lhalf = lane >> 4;       // lanes 0-15 vs 16-31

  const int blockN = blockIdx.x * BN;
  const int blockM = blockIdx.y * BM;

  // Global->LDS fill mapping: 128 rows x 32 bf16 (64B/row); thread owns half a row.
  const int ldRow = tid >> 1;           // 0..127
  const int ldCol = (tid & 1) * 16;     // 0 or 16

  auto issue = [&](int kt, int st) {
    const __bf16* gA = Xb + (size_t)(blockM + ldRow) * K_DIM + (size_t)kt * BK + ldCol;
    const __bf16* gB = Wb + (size_t)(blockN + ldRow) * K_DIM + (size_t)kt * BK + ldCol;
    cp16_g2l(gA,     &As[st][ldRow][ldCol]);
    cp16_g2l(gA + 8, &As[st][ldRow][ldCol + 8]);
    cp16_g2l(gB,     &Bs[st][ldRow][ldCol]);
    cp16_g2l(gB + 8, &Bs[st][ldRow][ldCol + 8]);
  };

  v8f zero = {};
  v8f acc[2][4];
#pragma unroll
  for (int i = 0; i < 2; ++i)
#pragma unroll
    for (int j = 0; j < 4; ++j) acc[i][j] = zero;

  constexpr int NT = K_DIM / BK;     // 128 K-tiles
  issue(0, 0);

  for (int kt = 0; kt < NT; ++kt) {
    const int st = kt & 1;
    wait_async_done();               // our stage-kt async fills are done
    __syncthreads();                 // all waves' fills visible; prev compute done
    if (kt + 1 < NT) issue(kt + 1, st ^ 1);  // overlap next fill with this WMMA

    // A fragments: lane l16 holds row m; K0..15 (lanes 0-15)
    // or K16..31 (lanes 16-31) -> two 16B LDS reads.
    v16bf afrag[2];
#pragma unroll
    for (int mt = 0; mt < 2; ++mt) {
      const __bf16* ap = &As[st][waveM * 32 + mt * 16 + l16][0];
      union { v16bf v; v8bf h[2]; } u;
      u.h[0] = *(const v8bf*)(ap + lhalf * 16);
      u.h[1] = *(const v8bf*)(ap + lhalf * 16 + 8);
      afrag[mt] = u.v;
    }
    // B fragments: lane l16 holds column n = W row n; lanes 0-15 take K0..15,
    // lanes 16-31 take K16..31 -> one contiguous 32B LDS read.
    v16bf bfrag[4];
#pragma unroll
    for (int nt = 0; nt < 4; ++nt) {
      const __bf16* bp = &Bs[st][waveN * 64 + nt * 16 + l16][lhalf * 16];
      bfrag[nt] = *(const v16bf*)bp;
    }

#pragma unroll
    for (int mt = 0; mt < 2; ++mt)
#pragma unroll
      for (int nt = 0; nt < 4; ++nt)
        acc[mt][nt] = __builtin_amdgcn_wmma_f32_16x16x32_bf16(
            false, afrag[mt], false, bfrag[nt],
            (short)0, acc[mt][nt], false, false);
  }

  // Epilogue: D VGPR r holds M = r (lanes 0-15) / M = 8+r (lanes 16-31), N = l16.
  float* outBase = Out + (size_t)(blockM + waveM * 32) * N_DIM + blockN + waveN * 64;
#pragma unroll
  for (int mt = 0; mt < 2; ++mt) {
#pragma unroll
    for (int nt = 0; nt < 4; ++nt) {
      const int col = nt * 16 + l16;
#pragma unroll
      for (int r = 0; r < 8; ++r) {
        const int row = mt * 16 + lhalf * 8 + r;
        outBase[(size_t)row * N_DIM + col] = acc[mt][nt][r];
      }
    }
  }
}

// ---- launcher ---------------------------------------------------------------
extern "C" void kernel_launch(void* const* d_in, const int* in_sizes, int n_in,
                              void* d_out, int out_size, void* d_ws, size_t ws_size,
                              hipStream_t stream) {
  const float*         x    = (const float*)d_in[0];
  const float*         w    = (const float*)d_in[1];
  const unsigned char* mask = (const unsigned char*)d_in[2];   // jnp bool = 1 byte
  float*               out  = (float*)d_out;

  // Workspace: Xb (64 MB) then Wb (32 MB) as bf16.
  __bf16* Xb = (__bf16*)d_ws;
  __bf16* Wb = Xb + (size_t)M_DIM * K_DIM;

  pack_x_kernel<<<(M_DIM * (size_t)K_DIM) / 8 / 256, 256, 0, stream>>>(x, Xb);
  pack_w_kernel<<<(N_DIM * (size_t)K_DIM) / 8 / 256, 256, 0, stream>>>(w, mask, Wb);

  dim3 grid(N_DIM / BN, M_DIM / BM);   // 32 x 64 = 2048 workgroups
  gemm_bf16_wmma_kernel<<<grid, NTHREADS, 0, stream>>>(Xb, Wb, out);
}
